// AFTBlock_90237262889181
// MI455X (gfx1250) — matrix operand order repaired and checked by reading, not compiled, not run on hardware
//
#include <hip/hip_runtime.h>
#include <hip/hip_bf16.h>

// ---------------------------------------------------------------------------
// AFT transformer block for MI455X (gfx1250, wave32, WMMA bf16 16x16x32,
// async global->LDS double-buffered B staging).
// ---------------------------------------------------------------------------

typedef __attribute__((ext_vector_type(16))) __bf16 v16bf;
typedef __attribute__((ext_vector_type(8)))  __bf16 v8bf;
typedef __attribute__((ext_vector_type(8)))  float  v8f;

#define BB 32
#define TT 512
#define DD 1024
#define FFD 4096
#define MTOT (BB * TT)   // 16384

union AFrag { v16bf v; v8bf h[2]; };

// Epilogue modes
#define EP_BIAS_BF16 0   // out(bf16) = acc + bias
#define EP_F32       1   // out(f32)  = acc
#define EP_RES_F32   2   // out(f32)  = acc + bias + resid
#define EP_GELU_BF16 3   // out(bf16) = gelu(acc + bias)   (exact erf)

// CDNA5 async copy: GLOBAL_LOAD_ASYNC_TO_LDS_B128 (ASYNCcnt-tracked).
// lds_off = raw DS byte offset (low 32 bits of the flat shared-memory address).
__device__ __forceinline__ void async_copy_b128(unsigned lds_off, const void* gaddr) {
  asm volatile("global_load_async_to_lds_b128 %0, %1, off"
               :: "v"(lds_off), "v"(gaddr) : "memory");
}
__device__ __forceinline__ void wait_asynccnt0() {
  asm volatile("s_wait_asynccnt 0" ::: "memory");
}
__device__ __forceinline__ unsigned lds_offset_of(const void* p) {
  return (unsigned)(unsigned long long)p;  // LDS aperture in upper 32 bits
}

// ---------------------------------------------------------------------------
// WMMA GEMM: C[M,N] = A[M,K](bf16,row-major) * Bt[N,K](bf16,N-major) (+epilogue)
// block = 256 threads (8 waves). Workgroup tile 256M x 64N, wave tile 32M x 64N.
// B tile (64N x 32K) double-buffered in LDS via async loads.
// Batched via blockIdx.z with element strides (aB, bB, cB).
// ---------------------------------------------------------------------------
template <int EPILOGUE>
__global__ __launch_bounds__(256) void gemm_wmma_kernel(
    const __bf16* __restrict__ A, const __bf16* __restrict__ Bt,
    void* __restrict__ Cv, const float* __restrict__ bias,
    const float* __restrict__ resid, int M, int N, int K,
    long long aB, long long bB, long long cB) {
  // Two LDS B tiles: 64 N-rows x 32 K each, padded to 48 elems/row
  // (96B rows, 32B-aligned) to spread DS banks on the fragment reads.
  __shared__ __align__(32) __bf16 Bs[2][64 * 48];

  const int tid  = threadIdx.x;
  const int lane = tid & 31;
  const int wv   = tid >> 5;       // wave id 0..7
  const int hl   = lane >> 4;      // half-wave 0/1
  const int l16  = lane & 15;

  const long long zb = blockIdx.z;
  A  += zb * aB;
  Bt += zb * bB;

  const int rowBase = blockIdx.y * 256 + wv * 32;  // wave's M origin (32 rows)
  const int colBase = blockIdx.x * 64;             // block's N origin

  v8f acc[2][4] = {};

  // Per-lane A row pointers (lane l16 owns rows rowBase+l16 and rowBase+16+l16)
  const __bf16* arow0 = A + (long long)(rowBase + l16) * K;
  const __bf16* arow1 = A + (long long)(rowBase + 16 + l16) * K;

  // Cooperative B-tile mapping: thread -> (n-row, 16B K-chunk); 256 x 16B = tile
  const int bn = tid >> 2;         // 0..63
  const int bc = tid & 3;          // 0..3 (chunk of 8 bf16)
  const __bf16* brow = Bt + (long long)(colBase + bn) * K + bc * 8;
  const unsigned ldsDst0 = lds_offset_of(&Bs[0][bn * 48 + bc * 8]);
  const unsigned ldsDst1 = lds_offset_of(&Bs[1][bn * 48 + bc * 8]);

  const int steps = K >> 5;  // K/32
  // Prologue: stage first B tile asynchronously
  async_copy_b128(ldsDst0, brow);

  for (int i = 0; i < steps; ++i) {
    const int k0 = i << 5;
    const int cur = i & 1;
    // Current tile resident; all waves' async writes drained + visible.
    wait_asynccnt0();
    __syncthreads();
    // Kick off next B tile into the other buffer (hazard-free: every wave's
    // LDS reads of that buffer completed before it passed the barrier above).
    if (i + 1 < steps)
      async_copy_b128(cur ? ldsDst0 : ldsDst1, brow + k0 + 32);

    // A fragments (contiguous K order, matching B):
    //   elems 0..7  = A[m][k0 + hl*16 + 0..7]
    //   elems 8..15 = A[m][k0 + hl*16 + 8..15]
    AFrag a0, a1;
    a0.h[0] = *(const v8bf*)(arow0 + k0 + hl * 16);
    a0.h[1] = *(const v8bf*)(arow0 + k0 + hl * 16 + 8);
    a1.h[0] = *(const v8bf*)(arow1 + k0 + hl * 16);
    a1.h[1] = *(const v8bf*)(arow1 + k0 + hl * 16 + 8);
    if (k0 + 32 < K) {
      __builtin_prefetch(arow0 + k0 + 32, 0, 3);  // global_prefetch_b8 (near)
      __builtin_prefetch(arow1 + k0 + 32, 0, 3);
    }

    const __bf16* bs = Bs[cur];
#pragma unroll
    for (int s = 0; s < 4; ++s) {
      // B fragment: lane owns column (s*16 + l16), K = hl*16 + e (contiguous)
      AFrag bf_;
      bf_.v = *(const v16bf*)(bs + (s * 16 + l16) * 48 + hl * 16);
      acc[0][s] = __builtin_amdgcn_wmma_f32_16x16x32_bf16(
          false, a0.v, false, bf_.v, (short)0, acc[0][s], false, false);
      acc[1][s] = __builtin_amdgcn_wmma_f32_16x16x32_bf16(
          false, a1.v, false, bf_.v, (short)0, acc[1][s], false, false);
    }
  }

  // Epilogue. C/D layout: lane l16 = column, VGPR r -> row hl*8 + r.
#pragma unroll
  for (int h2 = 0; h2 < 2; ++h2) {
    const int mBase = rowBase + h2 * 16 + hl * 8;
#pragma unroll
    for (int s = 0; s < 4; ++s) {
      const int col = colBase + s * 16 + l16;
      const float bval = (EPILOGUE == EP_F32) ? 0.0f : bias[col];
#pragma unroll
      for (int r = 0; r < 8; ++r) {
        const long long idx = cB * zb + (long long)(mBase + r) * N + col;
        float v = acc[h2][s][r] + bval;
        if (EPILOGUE == EP_BIAS_BF16) {
          ((__bf16*)Cv)[idx] = (__bf16)v;
        } else if (EPILOGUE == EP_F32) {
          ((float*)Cv)[idx] = v;
        } else if (EPILOGUE == EP_RES_F32) {
          ((float*)Cv)[idx] = v + resid[idx];
        } else {  // EP_GELU_BF16, exact erf GELU (torch nn.GELU default)
          v = 0.5f * v * (1.0f + erff(v * 0.70710678118654752f));
          ((__bf16*)Cv)[idx] = (__bf16)v;
        }
      }
    }
  }
}

// ---------------------------------------------------------------------------
// LayerNorm: one 256-thread block per row of [MTOT, DD]; bf16 output.
// wave32 shuffle reduction + 8-wave LDS combine.
// ---------------------------------------------------------------------------
__global__ __launch_bounds__(256) void layernorm_kernel(
    const float* __restrict__ x, const float* __restrict__ g,
    const float* __restrict__ be, __bf16* __restrict__ out) {
  __shared__ float red[8];
  const long long row = blockIdx.x;
  const float* xr = x + row * DD;
  const int tid = threadIdx.x;

  float v0 = xr[tid], v1 = xr[tid + 256], v2 = xr[tid + 512], v3 = xr[tid + 768];
  float s = v0 + v1 + v2 + v3;
#pragma unroll
  for (int o = 16; o > 0; o >>= 1) s += __shfl_down(s, o, 32);
  if ((tid & 31) == 0) red[tid >> 5] = s;
  __syncthreads();
  const float mean = (red[0] + red[1] + red[2] + red[3] + red[4] + red[5] +
                      red[6] + red[7]) * (1.0f / (float)DD);
  __syncthreads();

  const float d0 = v0 - mean, d1 = v1 - mean, d2 = v2 - mean, d3 = v3 - mean;
  float q = d0 * d0 + d1 * d1 + d2 * d2 + d3 * d3;
#pragma unroll
  for (int o = 16; o > 0; o >>= 1) q += __shfl_down(q, o, 32);
  if ((tid & 31) == 0) red[tid >> 5] = q;
  __syncthreads();
  const float var = (red[0] + red[1] + red[2] + red[3] + red[4] + red[5] +
                     red[6] + red[7]) * (1.0f / (float)DD);
  const float rinv = rsqrtf(var + 1e-5f);

  __bf16* orow = out + row * DD;
  orow[tid]       = (__bf16)(d0 * rinv * g[tid]       + be[tid]);
  orow[tid + 256] = (__bf16)(d1 * rinv * g[tid + 256] + be[tid + 256]);
  orow[tid + 512] = (__bf16)(d2 * rinv * g[tid + 512] + be[tid + 512]);
  orow[tid + 768] = (__bf16)(d3 * rinv * g[tid + 768] + be[tid + 768]);
}

// ---------------------------------------------------------------------------
// Weight prep: cast to bf16, transpose to N-major; fuse QKV; exp(wbias); bQKV.
// ---------------------------------------------------------------------------
__global__ void prep_kernel(const float* __restrict__ wq, const float* __restrict__ wk,
                            const float* __restrict__ wvp, const float* __restrict__ wp,
                            const float* __restrict__ w1, const float* __restrict__ w2,
                            const float* __restrict__ wbias,
                            const float* __restrict__ bq, const float* __restrict__ bk,
                            const float* __restrict__ bv,
                            __bf16* __restrict__ wqkvT, __bf16* __restrict__ wpT,
                            __bf16* __restrict__ w1T, __bf16* __restrict__ w2T,
                            __bf16* __restrict__ wbE, float* __restrict__ bQKV) {
  const long long N0 = 3072LL * 1024, N1 = 1024LL * 1024, N2 = 4096LL * 1024,
                  N3 = 1024LL * 4096, N4 = 512LL * 512, N5 = 3072;
  const long long total = N0 + N1 + N2 + N3 + N4 + N5;
  for (long long i = (long long)blockIdx.x * 256 + threadIdx.x; i < total;
       i += (long long)gridDim.x * 256) {
    if (i < N0) {
      long long n = i >> 10, k = i & 1023;
      float v = (n < 1024) ? wq[k * 1024 + n]
              : (n < 2048) ? wk[k * 1024 + (n - 1024)]
                           : wvp[k * 1024 + (n - 2048)];
      wqkvT[i] = (__bf16)v;
    } else if (i < N0 + N1) {
      long long j = i - N0, n = j >> 10, k = j & 1023;
      wpT[j] = (__bf16)wp[k * 1024 + n];
    } else if (i < N0 + N1 + N2) {
      long long j = i - N0 - N1, n = j >> 10, k = j & 1023;
      w1T[j] = (__bf16)w1[k * 4096 + n];
    } else if (i < N0 + N1 + N2 + N3) {
      long long j = i - N0 - N1 - N2, n = j >> 12, k = j & 4095;
      w2T[j] = (__bf16)w2[k * 1024 + n];
    } else if (i < N0 + N1 + N2 + N3 + N4) {
      long long j = i - N0 - N1 - N2 - N3;
      wbE[j] = (__bf16)expf(wbias[j]);
    } else {
      long long j = i - N0 - N1 - N2 - N3 - N4;
      bQKV[j] = (j < 1024) ? bq[j] : (j < 2048) ? bk[j - 1024] : bv[j - 2048];
    }
  }
}

// ---------------------------------------------------------------------------
// EVt pack: EVt[b][d][s]    = exp(K)*V   (bf16, s-contiguous for B^T of AFT GEMM)
//           EVt[b][D+d][s]  = exp(K)
// ---------------------------------------------------------------------------
__global__ __launch_bounds__(256) void aft_pack_kernel(
    const __bf16* __restrict__ QKV, __bf16* __restrict__ EVt) {
  long long gid = (long long)blockIdx.x * 256 + threadIdx.x;  // B*T*D threads
  long long b = gid / (TT * DD);
  long long r = gid % (TT * DD);
  long long s = r / DD, d = r % DD;
  const __bf16* base = QKV + (b * TT + s) * 3072;
  float kk = (float)base[1024 + d];
  float vv = (float)base[2048 + d];
  float ek = expf(kk);
  EVt[(b * 2048 + d)        * TT + s] = (__bf16)(ek * vv);
  EVt[(b * 2048 + 1024 + d) * TT + s] = (__bf16)ek;
}

// ---------------------------------------------------------------------------
// Gate: Y = sigmoid(Q) * num / den   (bf16 out for the wp projection GEMM)
// ---------------------------------------------------------------------------
__global__ __launch_bounds__(256) void aft_gate_kernel(
    const __bf16* __restrict__ QKV, const float* __restrict__ ND,
    __bf16* __restrict__ Yb) {
  long long gid = (long long)blockIdx.x * 256 + threadIdx.x;  // B*T*D threads
  long long b = gid / (TT * DD);
  long long r = gid % (TT * DD);
  long long t = r / DD, d = r % DD;
  float q   = (float)QKV[(b * TT + t) * 3072 + d];
  float num = ND[(b * TT + t) * 2048 + d];
  float den = ND[(b * TT + t) * 2048 + 1024 + d];
  float y = (1.0f / (1.0f + expf(-q))) * (num / den);
  Yb[(b * TT + t) * DD + d] = (__bf16)y;
}

// ---------------------------------------------------------------------------
// Host-side orchestration
// ---------------------------------------------------------------------------
extern "C" void kernel_launch(void* const* d_in, const int* in_sizes, int n_in,
                              void* d_out, int out_size, void* d_ws, size_t ws_size,
                              hipStream_t stream) {
  const float* x     = (const float*)d_in[0];
  const float* gamma = (const float*)d_in[1];
  const float* beta  = (const float*)d_in[2];
  const float* wq    = (const float*)d_in[3];
  const float* bq    = (const float*)d_in[4];
  const float* wk    = (const float*)d_in[5];
  const float* bk    = (const float*)d_in[6];
  const float* wvp   = (const float*)d_in[7];
  const float* bv    = (const float*)d_in[8];
  const float* wp    = (const float*)d_in[9];
  const float* bp    = (const float*)d_in[10];
  const float* wbias = (const float*)d_in[11];
  const float* w1    = (const float*)d_in[12];
  const float* b1    = (const float*)d_in[13];
  const float* w2    = (const float*)d_in[14];
  const float* b2    = (const float*)d_in[15];
  float* out = (float*)d_out;

  // Workspace carve-up (256B aligned)
  char* ws = (char*)d_ws;
  size_t off = 0;
  auto alloc = [&](size_t bytes) -> void* {
    void* p = ws + off;
    off = (off + bytes + 255) & ~(size_t)255;
    return p;
  };
  __bf16* wqkvT = (__bf16*)alloc(3072ULL * 1024 * 2);
  __bf16* wpT   = (__bf16*)alloc(1024ULL * 1024 * 2);
  __bf16* w1T   = (__bf16*)alloc(4096ULL * 1024 * 2);
  __bf16* w2T   = (__bf16*)alloc(1024ULL * 4096 * 2);
  __bf16* wbE   = (__bf16*)alloc(512ULL * 512 * 2);
  float*  bQKV  = (float*)alloc(3072ULL * 4);
  __bf16* hB    = (__bf16*)alloc((size_t)MTOT * DD * 2);     // LN output (reused)
  __bf16* QKVb  = (__bf16*)alloc((size_t)MTOT * 3072 * 2);   // [B,T,3D]
  __bf16* EVt   = (__bf16*)alloc((size_t)BB * 2048 * TT * 2);// [B,2D,T]
  float*  ND    = (float*)alloc((size_t)BB * TT * 2048 * 4); // [B,T,2D] num|den
  __bf16* Yb    = (__bf16*)alloc((size_t)MTOT * DD * 2);     // gated attn out
  float*  x1    = (float*)alloc((size_t)MTOT * DD * 4);      // residual 1
  __bf16* G     = (__bf16*)alloc((size_t)MTOT * FFD * 2);    // gelu(h@w1+b1)
  (void)ws_size; (void)in_sizes; (void)n_in; (void)out_size;

  const long long nElem = (long long)BB * TT * DD;  // 16.7M

  // 1) weight prep (bf16 transpose / fuse / exp(wbias))
  prep_kernel<<<8192, 256, 0, stream>>>(wq, wk, wvp, wp, w1, w2, wbias, bq, bk, bv,
                                        wqkvT, wpT, w1T, w2T, wbE, bQKV);
  // 2) LN1: h = LN(x)
  layernorm_kernel<<<MTOT, 256, 0, stream>>>(x, gamma, beta, hB);
  // 3) QKV = h @ [wq|wk|wv] + [bq|bk|bv]   (M=16384, K=1024, N=3072)
  gemm_wmma_kernel<EP_BIAS_BF16><<<dim3(3072 / 64, MTOT / 256, 1), 256, 0, stream>>>(
      hB, wqkvT, QKVb, bQKV, nullptr, MTOT, 3072, DD, 0, 0, 0);
  // 4) pack exp(K)*V | exp(K), transposed per batch
  aft_pack_kernel<<<(unsigned)(nElem / 256), 256, 0, stream>>>(QKVb, EVt);
  // 5) [num|den] = exp(wbias) @ EVt^T   (batched: M=512, K=512, N=2048, 32 batches)
  gemm_wmma_kernel<EP_F32><<<dim3(2048 / 64, TT / 256, BB), 256, 0, stream>>>(
      wbE, EVt, ND, nullptr, nullptr, TT, 2048, TT,
      0LL, 2048LL * TT, (long long)TT * 2048);
  // 6) Y = sigmoid(Q) * num / den
  aft_gate_kernel<<<(unsigned)(nElem / 256), 256, 0, stream>>>(QKVb, ND, Yb);
  // 7) x1 = x + Y @ wp + bp   (M=16384, K=1024, N=1024)
  gemm_wmma_kernel<EP_RES_F32><<<dim3(1024 / 64, MTOT / 256, 1), 256, 0, stream>>>(
      Yb, wpT, x1, bp, x, MTOT, DD, DD, 0, 0, 0);
  // 8) LN2: h = LN(x1)
  layernorm_kernel<<<MTOT, 256, 0, stream>>>(x1, gamma, beta, hB);
  // 9) G = gelu(h @ w1 + b1)   (M=16384, K=1024, N=4096)
  gemm_wmma_kernel<EP_GELU_BF16><<<dim3(FFD / 64, MTOT / 256, 1), 256, 0, stream>>>(
      hB, w1T, G, b1, nullptr, MTOT, FFD, DD, 0, 0, 0);
  // 10) out = x1 + G @ w2 + b2   (M=16384, K=4096, N=1024)
  gemm_wmma_kernel<EP_RES_F32><<<dim3(1024 / 64, MTOT / 256, 1), 256, 0, stream>>>(
      G, w2T, out, b2, x1, MTOT, DD, FFD, 0, 0, 0);
}
